// TitanMini_60816736911470
// MI455X (gfx1250) — matrix operand order repaired and checked by reading, not compiled, not run on hardware
//
#include <hip/hip_runtime.h>
#include <hip/hip_bf16.h>
#include <math.h>

// ---------------------------------------------------------------------------
// CDNA5 (gfx1250) wave32 WMMA TitanMini forward.
//  - all GEMMs: v_wmma_f32_16x16x32_bf16, bf16 activations + bf16 weights
//    (104MB of bf16 weights fit in the 192MB L2 -> compute bound)
//  - A tiles staged via async global->LDS copies (ASYNCcnt, gfx1250 path)
//  - double-buffered, software-pipelined K loop
// ---------------------------------------------------------------------------

typedef __attribute__((ext_vector_type(16))) __bf16 v16bf;
typedef __attribute__((ext_vector_type(8)))  float  v8f;
typedef int v4i_vs __attribute__((vector_size(16)));   // matches builtin param

#define GTM 128   // block tile M
#define GTN 64    // block tile N
#define GTK 32    // K step (16x16x32 WMMA)
#define SA_STRIDE 40  // bf16 elems per LDS A row: 80B = 20 banks, 16B aligned
#define SB_STRIDE 40  // bf16 elems per LDS B column (stored transposed)

#ifndef __has_builtin
#define __has_builtin(x) 0
#endif
#if __has_builtin(__builtin_amdgcn_global_load_async_to_lds_b128)
#define HAVE_ASYNC_LDS 1
typedef __attribute__((address_space(1))) v4i_vs* gbl4p;
typedef __attribute__((address_space(3))) v4i_vs* lds4p;
#else
#define HAVE_ASYNC_LDS 0
#endif

__device__ __forceinline__ void wait_async0() {
#if HAVE_ASYNC_LDS
# if __has_builtin(__builtin_amdgcn_s_wait_asynccnt)
  __builtin_amdgcn_s_wait_asynccnt(0);
# else
  asm volatile("s_wait_asynccnt 0x0" ::: "memory");
# endif
#endif
}

// ------------------------------ weight convert -----------------------------
__global__ void cvt_bf16_kernel(const float* __restrict__ src,
                                __bf16* __restrict__ dst, int n) {
  int i = blockIdx.x * 256 + threadIdx.x;
  if (i < n) dst[i] = (__bf16)src[i];
}

// ------------------------------ WMMA GEMM ----------------------------------
// Out = epi( A[M,K](bf16) x W[K,N](bf16) + bias ) (+ resid)
// Requirements (all callers satisfy): M%128==0, K%32==0, lda/ldw%8==0.
// N edge handled (policy head N=72). epi: 0 none, 1 exact GELU.
__global__ __launch_bounds__(256) void gemm_bf16_wmma(
    const __bf16* __restrict__ A, int lda,
    const __bf16* __restrict__ W, int ldw,
    const float* __restrict__ bias,
    const float* __restrict__ resid, int ldr,
    float* __restrict__ OutF, __bf16* __restrict__ OutB,
    int ldc, int M, int N, int K, int epi)
{
  __shared__ __bf16 sA[2][GTM * SA_STRIDE];   // [row][k]
  __shared__ __bf16 sB[2][GTN * SB_STRIDE];   // transposed: [col][k]

  const int tid  = threadIdx.x;
  const int lane = tid & 31;
  const int wave = tid >> 5;
  const int wm   = (wave >> 1) * 32;
  const int wn   = (wave & 1) * 32;
  const int half = lane >> 4;
  const int r    = lane & 15;
  const int bm   = blockIdx.y * GTM;
  const int bn   = blockIdx.x * GTN;

  // staging coordinates (per thread)
  const int arow = tid >> 2;            // 0..63 (+64 for second chunk)
  const int ach  = (tid & 3) * 8;       // bf16 elems; 16B chunk in 64B row
  const int bk   = tid >> 3;            // 0..31
  const int bcc  = (tid & 7) * 8;       // 8-col chunk

  uint4 a0, a1, bv;                     // register staging (sync fallback / B)

  auto issueA = [&](int k0g, __bf16* dst) {
#if HAVE_ASYNC_LDS
    #pragma unroll
    for (int it = 0; it < 2; ++it) {
      const __bf16* g = A + (size_t)(bm + arow + 64 * it) * lda + k0g + ach;
      __bf16* l = dst + (arow + 64 * it) * SA_STRIDE + ach;
      __builtin_amdgcn_global_load_async_to_lds_b128(
          (gbl4p)(uintptr_t)g,
          (lds4p)(uint32_t)(uintptr_t)l,
          0, 0);
    }
#else
    (void)k0g; (void)dst;
#endif
  };
  auto loadA = [&](int k0g) {
#if !HAVE_ASYNC_LDS
    a0 = *(const uint4*)(A + (size_t)(bm + arow) * lda + k0g + ach);
    a1 = *(const uint4*)(A + (size_t)(bm + arow + 64) * lda + k0g + ach);
#else
    (void)k0g;
#endif
  };
  auto storeA = [&](__bf16* dst) {
#if !HAVE_ASYNC_LDS
    *(uint4*)(dst + arow * SA_STRIDE + ach) = a0;
    *(uint4*)(dst + (arow + 64) * SA_STRIDE + ach) = a1;
#else
    (void)dst;
#endif
  };
  auto loadB = [&](int k0g) {
    bv = *(const uint4*)(W + (size_t)(k0g + bk) * ldw + bn + bcc);
  };
  auto storeB = [&](__bf16* dst) {
    const __bf16* e = (const __bf16*)&bv;
    #pragma unroll
    for (int i = 0; i < 8; ++i) dst[(bcc + i) * SB_STRIDE + bk] = e[i];
  };

  v8f zero = {0.f,0.f,0.f,0.f,0.f,0.f,0.f,0.f};
  v8f accs[2][2];
  accs[0][0] = zero; accs[0][1] = zero; accs[1][0] = zero; accs[1][1] = zero;

  // prologue: stage K-step 0 into buffer 0
  issueA(0, sA[0]); loadA(0); loadB(0); storeA(sA[0]); storeB(sB[0]);
  wait_async0();
  __syncthreads();

  const int ksteps = K >> 5;
  for (int ks = 0; ks < ksteps; ++ks) {
    const int cur = ks & 1;
    const bool more = (ks + 1 < ksteps);
    const int k1 = (ks + 1) * GTK;
    if (more) { issueA(k1, sA[cur ^ 1]); loadA(k1); loadB(k1); }

    // ---- pack fragments per ISA 16-bit A(16x32)/B(32x16) layouts ----
    const __bf16* sAc = sA[cur];
    const __bf16* sBc = sB[cur];
    v16bf afr[2], bfr[2];
    #pragma unroll
    for (int jj = 0; jj < 8; ++jj) {
      const int ka = ((jj < 4) ? (2 * jj) : (2 * jj + 8)) + 8 * half;
      #pragma unroll
      for (int i = 0; i < 2; ++i) {
        const int rb = (wm + 16 * i + r) * SA_STRIDE;
        afr[i][2 * jj]     = sAc[rb + ka];
        afr[i][2 * jj + 1] = sAc[rb + ka + 1];
      }
      const int kb2 = ka;
      #pragma unroll
      for (int j = 0; j < 2; ++j) {
        const int cb = (wn + 16 * j + r) * SB_STRIDE;
        bfr[j][2 * jj]     = sBc[cb + kb2];
        bfr[j][2 * jj + 1] = sBc[cb + kb2 + 1];
      }
    }
    #pragma unroll
    for (int i = 0; i < 2; ++i)
      #pragma unroll
      for (int j = 0; j < 2; ++j)
        accs[i][j] = __builtin_amdgcn_wmma_f32_16x16x32_bf16(
            false, afr[i], false, bfr[j], (short)0, accs[i][j], false, false);

    if (more) { storeA(sA[cur ^ 1]); storeB(sB[cur ^ 1]); }
    wait_async0();
    __syncthreads();
  }

  // ---- epilogue: C layout: VGPR e, lane -> (row e + 8*half, col r) ----
  const bool nfull = (bn + GTN <= N);
  #pragma unroll
  for (int i = 0; i < 2; ++i) {
    #pragma unroll
    for (int j = 0; j < 2; ++j) {
      #pragma unroll
      for (int e = 0; e < 8; ++e) {
        const int gr = bm + wm + 16 * i + e + 8 * half;
        const int gc = bn + wn + 16 * j + r;
        if (nfull || gc < N) {
          float vo = accs[i][j][e] + bias[gc];
          if (epi == 1) vo = 0.5f * vo * (1.f + erff(vo * 0.70710678f));
          if (resid) vo += resid[(size_t)gr * ldr + gc];
          if (OutF) OutF[(size_t)gr * ldc + gc] = vo;
          if (OutB) OutB[(size_t)gr * ldc + gc] = (__bf16)vo;
        }
      }
    }
  }
}

// ------------------------------ LayerNorm (f32 in, bf16 out) ---------------
__global__ __launch_bounds__(256) void ln_kernel(
    const float* __restrict__ X, const float* __restrict__ g,
    const float* __restrict__ b, __bf16* __restrict__ Y, int T)
{
  const int tok = blockIdx.x * 8 + (threadIdx.x >> 5);
  if (tok >= T) return;
  const int lane = threadIdx.x & 31;
  const float* xr = X + (size_t)tok * 512;
  float rv[16], s = 0.f, sq = 0.f;
  #pragma unroll
  for (int i = 0; i < 16; ++i) {
    float x = xr[lane + 32 * i];
    rv[i] = x; s += x; sq += x * x;
  }
  #pragma unroll
  for (int o = 16; o >= 1; o >>= 1) {
    s += __shfl_xor(s, o, 32); sq += __shfl_xor(sq, o, 32);
  }
  const float mean = s * (1.f / 512.f);
  const float var  = sq * (1.f / 512.f) - mean * mean;
  const float rstd = rsqrtf(var + 1e-5f);
  __bf16* yr = Y + (size_t)tok * 512;
  #pragma unroll
  for (int i = 0; i < 16; ++i) {
    const int d = lane + 32 * i;
    yr[d] = (__bf16)((rv[i] - mean) * rstd * g[d] + b[d]);
  }
}

// ------------------------------ Attention (bf16 qkv -> bf16 ctx) -----------
__global__ __launch_bounds__(256) void attn_kernel(
    const __bf16* __restrict__ Q, const __bf16* __restrict__ Kp,
    const __bf16* __restrict__ V, const float* __restrict__ rel,
    __bf16* __restrict__ ctx)
{
  __shared__ float sQ[65 * 64];
  __shared__ float sK[65 * 65];   // stride 65 -> conflict-free column reads
  __shared__ float sV[65 * 65];
  __shared__ float sP[8][72];

  const int bh = blockIdx.x;
  const int b = bh >> 3, hh = bh & 7;
  const int tid = threadIdx.x, lane = tid & 31, wave = tid >> 5;
  const size_t base = ((size_t)b * 65) * 512 + (size_t)hh * 64;

  for (int i = tid; i < 65 * 64; i += 256) {
    const int row = i >> 6, d = i & 63;
    const size_t gidx = base + (size_t)row * 512 + d;
    sQ[row * 64 + d] = (float)Q[gidx];
    sK[row * 65 + d] = (float)Kp[gidx];
    sV[row * 65 + d] = (float)V[gidx];
  }
  __syncthreads();

  const float scale = 0.125f;  // 1/sqrt(64)
  for (int t = wave; t < 65; t += 8) {
    float sc[3] = {-1e30f, -1e30f, -1e30f};
    #pragma unroll
    for (int u = 0; u < 3; ++u) {
      const int s = lane + 32 * u;
      if (s < 65) {
        float acc = 0.f;
        for (int d = 0; d < 64; ++d) acc += sQ[t * 64 + d] * sK[s * 65 + d];
        acc *= scale;
        if (t >= 1 && s >= 1) {
          const int qs = t - 1, kk = s - 1;
          int dr = (qs >> 3) - (kk >> 3);
          dr = (dr < -7 ? -7 : (dr > 7 ? 7 : dr)) + 7;
          int df = (qs & 7) - (kk & 7);
          df = (df < -7 ? -7 : (df > 7 ? 7 : df)) + 7;
          acc += rel[(dr * 15 + df) * 8 + hh];
        }
        sc[u] = acc;
      }
    }
    float m = fmaxf(sc[0], fmaxf(sc[1], sc[2]));
    #pragma unroll
    for (int o = 16; o >= 1; o >>= 1) m = fmaxf(m, __shfl_xor(m, o, 32));
    float e[3], sum = 0.f;
    #pragma unroll
    for (int u = 0; u < 3; ++u) {
      e[u] = __expf(sc[u] - m);
      sum += (lane + 32 * u < 65) ? e[u] : 0.f;
    }
    #pragma unroll
    for (int o = 16; o >= 1; o >>= 1) sum += __shfl_xor(sum, o, 32);
    const float inv = 1.f / sum;
    #pragma unroll
    for (int u = 0; u < 3; ++u) {
      const int s = lane + 32 * u;
      if (s < 65) sP[wave][s] = e[u] * inv;
    }
    #pragma unroll
    for (int u = 0; u < 2; ++u) {
      const int d = lane + 32 * u;
      float acc = 0.f;
      for (int s = 0; s < 65; ++s) acc += sP[wave][s] * sV[s * 65 + d];
      ctx[base + (size_t)t * 512 + d] = (__bf16)acc;
    }
  }
}

// ------------------------------ elementwise glue ---------------------------
__global__ void gelu_gate_kernel(const __bf16* __restrict__ ab,
                                 __bf16* __restrict__ outp, int T) {
  const size_t i = (size_t)blockIdx.x * 256 + threadIdx.x;
  if (i >= (size_t)T * 1920) return;
  const size_t t = i / 1920;
  const int j = (int)(i % 1920);
  const float a = (float)ab[t * 3840 + j];
  const float gg = (float)ab[t * 3840 + 1920 + j];
  const float ge = 0.5f * a * (1.f + erff(a * 0.70710678f));
  outp[i] = (__bf16)(ge * gg);
}

__global__ __launch_bounds__(256) void embed_kernel(
    const float* __restrict__ x, const float* __restrict__ conv_w,
    const float* __restrict__ conv_b, const float* __restrict__ clsw,
    const float* __restrict__ abs_pos, const float* __restrict__ file_emb,
    const float* __restrict__ rank_emb, const float* __restrict__ diag_emb,
    const float* __restrict__ anti_emb, float* __restrict__ h)
{
  __shared__ float xs[112 * 64];
  const int b = blockIdx.x, tid = threadIdx.x;
  for (int i = tid; i < 112 * 64; i += 256) xs[i] = x[(size_t)b * 7168 + i];
  __syncthreads();
  for (int o = tid; o < 64 * 512; o += 256) {
    const int s = o & 63, d = o >> 6;
    const int f = s & 7, rk = s >> 3;
    float acc = conv_b[d] + abs_pos[s * 512 + d] + file_emb[f * 512 + d]
              + rank_emb[rk * 512 + d] + diag_emb[(rk + f) * 512 + d]
              + anti_emb[(rk - f + 7) * 512 + d];
    for (int c = 0; c < 112; ++c) acc += xs[c * 64 + s] * conv_w[d * 112 + c];
    h[((size_t)b * 65 + 1 + s) * 512 + d] = acc;
  }
  for (int d = tid; d < 512; d += 256) h[((size_t)b * 65) * 512 + d] = clsw[d];
}

__global__ void gather_kernel(const __bf16* __restrict__ y,
                              __bf16* __restrict__ board,
                              __bf16* __restrict__ clsb) {
  const size_t i = (size_t)blockIdx.x * 256 + threadIdx.x;
  if (i >= (size_t)66560 * 512) return;
  const int d = (int)(i & 511);
  const int row = (int)(i >> 9);
  const int b = row / 65, s = row - b * 65;
  if (s == 0) clsb[(size_t)b * 512 + d] = y[i];
  else board[((size_t)(b * 64 + s - 1)) * 512 + d] = y[i];
}

__global__ void pol_scatter_kernel(const float* __restrict__ pr,
                                   float* __restrict__ outp) {
  const int i = blockIdx.x * 256 + threadIdx.x;
  if (i >= 1024 * 64 * 72) return;
  const int p = i % 72, bs = i / 72;
  const int s = bs & 63, b = bs >> 6;
  outp[(size_t)b * 4608 + p * 64 + s] = pr[i];
}

__global__ __launch_bounds__(32) void value_head_kernel(
    const float* __restrict__ t, const float* __restrict__ vs_w,
    const float* __restrict__ vs_b, const float* __restrict__ vw_w,
    const float* __restrict__ vw_b, float* __restrict__ scalar_out,
    float* __restrict__ wdl_out)
{
  const int b = blockIdx.x, lane = threadIdx.x;
  const float* tr = t + (size_t)b * 256;
  float a0 = 0.f, a1 = 0.f, a2 = 0.f, a3 = 0.f;
  for (int j = lane; j < 256; j += 32) {
    const float tv = tr[j];
    a0 += tv * vs_w[j];
    a1 += tv * vw_w[j * 3 + 0];
    a2 += tv * vw_w[j * 3 + 1];
    a3 += tv * vw_w[j * 3 + 2];
  }
  #pragma unroll
  for (int o = 16; o >= 1; o >>= 1) {
    a0 += __shfl_xor(a0, o, 32); a1 += __shfl_xor(a1, o, 32);
    a2 += __shfl_xor(a2, o, 32); a3 += __shfl_xor(a3, o, 32);
  }
  if (lane == 0) {
    scalar_out[b] = tanhf(a0 + vs_b[0]);
    wdl_out[b * 3 + 0] = a1 + vw_b[0];
    wdl_out[b * 3 + 1] = a2 + vw_b[1];
    wdl_out[b * 3 + 2] = a3 + vw_b[2];
  }
}

// ---------------------------------------------------------------------------
extern "C" void kernel_launch(void* const* d_in, const int* in_sizes, int n_in,
                              void* d_out, int out_size, void* d_ws, size_t ws_size,
                              hipStream_t stream) {
  (void)in_sizes; (void)n_in; (void)out_size; (void)ws_size;
  // jax pytree flatten order (dicts alphabetical, lists in order):
  // 0 abs_pos, 1 anti_emb, 2 cls, 3 conv_b, 4 conv_w, 5 diag_emb, 6 file_emb,
  // 7..227 layers (17 each: bk,bo,bq,bv,ff_b,ff_w,fo_b,fo_w,ln1_b,ln1_g,
  //                 ln2_b,ln2_g,rel,wk,wo,wq,wv),
  // 228 onorm_b, 229 onorm_g, 230 pol_b, 231 pol_w, 232 rank_emb,
  // 233 vs_b, 234 vs_w, 235 vt_b, 236 vt_w, 237 vw_b, 238 vw_w, 239 x
  auto P = [&](int i) { return (const float*)d_in[i]; };
  const int Ttok = 66560;  // 1024 * 65

  char* ws = (char*)d_ws;
  size_t off = 0;
  auto AL = [&](size_t bytes) -> char* {
    char* p = ws + off;
    off = (off + bytes + 255) & ~(size_t)255;
    return p;
  };
  __bf16 *bwq[13], *bwk[13], *bwv[13], *bwo[13], *bffw[13], *bfow[13];
  for (int l = 0; l < 13; ++l) {
    bwq[l]  = (__bf16*)AL((size_t)512 * 512 * 2);
    bwk[l]  = (__bf16*)AL((size_t)512 * 512 * 2);
    bwv[l]  = (__bf16*)AL((size_t)512 * 512 * 2);
    bwo[l]  = (__bf16*)AL((size_t)512 * 512 * 2);
    bffw[l] = (__bf16*)AL((size_t)512 * 3840 * 2);
    bfow[l] = (__bf16*)AL((size_t)1920 * 512 * 2);
  }
  __bf16* bpolw = (__bf16*)AL((size_t)512 * 72 * 2);
  __bf16* bvtw  = (__bf16*)AL((size_t)512 * 256 * 2);
  float*  h     = (float*)AL((size_t)Ttok * 512 * 4);
  __bf16* y     = (__bf16*)AL((size_t)Ttok * 512 * 2);
  __bf16* qb    = (__bf16*)AL((size_t)Ttok * 512 * 2);
  __bf16* kb    = (__bf16*)AL((size_t)Ttok * 512 * 2);
  __bf16* vb    = (__bf16*)AL((size_t)Ttok * 512 * 2);
  __bf16* ctxb  = (__bf16*)AL((size_t)Ttok * 512 * 2);
  __bf16* ab    = (__bf16*)AL((size_t)Ttok * 3840 * 2);
  __bf16* ga    = (__bf16*)AL((size_t)Ttok * 1920 * 2);
  __bf16* board = (__bf16*)AL((size_t)65536 * 512 * 2);
  __bf16* clsb  = (__bf16*)AL((size_t)1024 * 512 * 2);
  float* polraw = (float*)AL((size_t)65536 * 72 * 4);
  float* tbuf   = (float*)AL((size_t)1024 * 256 * 4);

  auto cvt = [&](const float* s, __bf16* d, int n) {
    cvt_bf16_kernel<<<(n + 255) / 256, 256, 0, stream>>>(s, d, n);
  };
  auto gemm = [&](const __bf16* A, int lda, const __bf16* W, int ldw,
                  const float* bias, const float* resid, int ldr,
                  float* OutF, __bf16* OutB, int ldc,
                  int M, int N, int K, int epi) {
    dim3 g((N + GTN - 1) / GTN, (M + GTM - 1) / GTM);
    gemm_bf16_wmma<<<g, 256, 0, stream>>>(A, lda, W, ldw, bias, resid, ldr,
                                          OutF, OutB, ldc, M, N, K, epi);
  };

  // ---- convert all GEMM weights to bf16 (L2-resident working set) ----
  for (int l = 0; l < 13; ++l) {
    const int bl = 7 + l * 17;
    cvt(P(bl + 15), bwq[l], 512 * 512);   // wq
    cvt(P(bl + 13), bwk[l], 512 * 512);   // wk
    cvt(P(bl + 16), bwv[l], 512 * 512);   // wv
    cvt(P(bl + 14), bwo[l], 512 * 512);   // wo
    cvt(P(bl + 5),  bffw[l], 512 * 3840); // ff_w
    cvt(P(bl + 7),  bfow[l], 1920 * 512); // fo_w
  }
  cvt(P(231), bpolw, 512 * 72);
  cvt(P(236), bvtw, 512 * 256);

  // ---- embedding ----
  embed_kernel<<<1024, 256, 0, stream>>>(P(239), P(4), P(3), P(2), P(0), P(6),
                                         P(232), P(5), P(1), h);

  // ---- transformer layers ----
  const int lnGrid = (Ttok + 7) / 8;
  for (int l = 0; l < 13; ++l) {
    const int bl = 7 + l * 17;
    ln_kernel<<<lnGrid, 256, 0, stream>>>(h, P(bl + 9), P(bl + 8), y, Ttok);
    gemm(y, 512, bwq[l], 512, P(bl + 2), nullptr, 0, nullptr, qb, 512, Ttok, 512, 512, 0);
    gemm(y, 512, bwk[l], 512, P(bl + 0), nullptr, 0, nullptr, kb, 512, Ttok, 512, 512, 0);
    gemm(y, 512, bwv[l], 512, P(bl + 3), nullptr, 0, nullptr, vb, 512, Ttok, 512, 512, 0);
    attn_kernel<<<1024 * 8, 256, 0, stream>>>(qb, kb, vb, P(bl + 12), ctxb);
    gemm(ctxb, 512, bwo[l], 512, P(bl + 1), h, 512, h, nullptr, 512, Ttok, 512, 512, 0);
    ln_kernel<<<lnGrid, 256, 0, stream>>>(h, P(bl + 11), P(bl + 10), y, Ttok);
    gemm(y, 512, bffw[l], 3840, P(bl + 4), nullptr, 0, nullptr, ab, 3840, Ttok, 3840, 512, 0);
    gelu_gate_kernel<<<(int)(((size_t)Ttok * 1920 + 255) / 256), 256, 0, stream>>>(ab, ga, Ttok);
    gemm(ga, 1920, bfow[l], 512, P(bl + 6), h, 512, h, nullptr, 512, Ttok, 512, 1920, 0);
  }

  // ---- heads ----
  ln_kernel<<<lnGrid, 256, 0, stream>>>(h, P(229), P(228), y, Ttok);
  gather_kernel<<<(int)(((size_t)Ttok * 512 + 255) / 256), 256, 0, stream>>>(y, board, clsb);
  gemm(board, 512, bpolw, 72, P(230), nullptr, 0, polraw, nullptr, 72, 65536, 72, 512, 0);
  pol_scatter_kernel<<<(1024 * 64 * 72 + 255) / 256, 256, 0, stream>>>(polraw, (float*)d_out);
  gemm(clsb, 512, bvtw, 256, P(235), nullptr, 0, tbuf, nullptr, 256, 1024, 256, 512, 1);
  value_head_kernel<<<1024, 32, 0, stream>>>(tbuf, P(234), P(233), P(238), P(237),
                                             (float*)d_out + 4718592,
                                             (float*)d_out + 4719616);
}
